// FlattenedVectorQuantizerEMA_10239202034430
// MI455X (gfx1250) — matrix operand order repaired and checked by reading, not compiled, not run on hardware
//
#include <hip/hip_runtime.h>
#include <cstddef>

// ---------------- problem constants ----------------
#define N_TOKENS 131072
#define CODE_DIM 256
#define K_CODES  1024
#define DECAY    0.99f
#define ONE_M_D  0.01f
#define COMMIT   0.25f
#define EPSV     1e-5f

// ---------------- vector types ----------------
typedef __attribute__((ext_vector_type(16))) __bf16          v16bf;
typedef __attribute__((ext_vector_type(8)))  float           v8f;
typedef __attribute__((ext_vector_type(16))) unsigned short  v16us;
typedef __attribute__((ext_vector_type(4)))  unsigned short  v4us;

// ---------------- workspace layout (bytes) ----------------
// [dw 1MB][counts 4KB][lossSum 4][nSum 4][pad][idx 512KB][ebh 512KB][ebl 512KB][enorm 4KB]
static constexpr size_t WS_DW     = 0;
static constexpr size_t WS_COUNTS = WS_DW + (size_t)K_CODES * CODE_DIM * 4;   // 1048576
static constexpr size_t WS_LOSS   = WS_COUNTS + (size_t)K_CODES * 4;          // 1052672
static constexpr size_t WS_NSUM   = WS_LOSS + 4;                              // 1052676
static constexpr size_t WS_IDX    = 1052928;                                  // aligned
static constexpr size_t WS_EBH    = WS_IDX + (size_t)N_TOKENS * 4;            // 1577216
static constexpr size_t WS_EBL    = WS_EBH + (size_t)K_CODES * CODE_DIM * 2;  // 2101504
static constexpr size_t WS_ENORM  = WS_EBL + (size_t)K_CODES * CODE_DIM * 2;  // 2625792
static constexpr int    ZERO_FLOATS = (int)((WS_NSUM + 4) / 4);               // dw+counts+scalars

// ---------------- helpers ----------------
__device__ __forceinline__ void bf16_split(float x, unsigned short& hi, unsigned short& lo) {
    unsigned int b = __float_as_uint(x);
    hi = (unsigned short)(b >> 16);                    // truncating chop -> exact residual
    float r = x - __uint_as_float((unsigned int)hi << 16);
    lo = (unsigned short)(__float_as_uint(r) >> 16);
}

// ---------------- kernel 1: zero scratch ----------------
__global__ void zero_ws_kernel(float* __restrict__ p, int n) {
    int i = blockIdx.x * blockDim.x + threadIdx.x;
    if (i < n) p[i] = 0.0f;
}

// ---------------- kernel 2: split codebook to bf16 hi/lo + norms ----------------
__global__ void prep_embed_kernel(const float* __restrict__ embed,
                                  unsigned short* __restrict__ ebh,
                                  unsigned short* __restrict__ ebl,
                                  float* __restrict__ enorm) {
    int k = blockIdx.x, d = threadIdx.x;
    size_t off = (size_t)k * CODE_DIM + d;
    float x = embed[off];
    unsigned short hi, lo;
    bf16_split(x, hi, lo);
    ebh[off] = hi;
    ebl[off] = lo;
    float s = x * x;
    for (int m = 16; m; m >>= 1) s += __shfl_xor(s, m, 32);
    __shared__ float parts[8];
    if ((threadIdx.x & 31) == 0) parts[threadIdx.x >> 5] = s;
    __syncthreads();
    if (threadIdx.x == 0) {
        float t = 0.f;
        #pragma unroll
        for (int i = 0; i < 8; ++i) t += parts[i];
        enorm[k] = t;
    }
}

// ---------------- kernel 3: WMMA distance + argmin ----------------
// 4 waves per block; one 16-token tile per wave.
// LDS per wave: 4096 ushort (A-hi fragments) + 4096 ushort (A-lo) = 16KB.
__global__ __launch_bounds__(128) void vq_argmin_kernel(
        const float* __restrict__ z,
        const unsigned short* __restrict__ ebh,
        const unsigned short* __restrict__ ebl,
        const float* __restrict__ enorm,
        int*   __restrict__ idx_out,
        float* __restrict__ idx_out_f) {
    __shared__ unsigned short lds[4 * 8192];   // 64KB
    const int lane = threadIdx.x & 31;
    const int wave = threadIdx.x >> 5;
    const int tile = blockIdx.x * 4 + wave;
    const int m0   = tile * 16;
    unsigned short* aw = lds + wave * 8192;    // hi at +0, lo at +4096

    // ---- stage A tile (16 x 256 f32) into LDS in WMMA A-fragment order ----
    // A order used here (contiguous k, B matches): lanes 0-15 hold K 0-15 (elem 0-15),
    // lanes 16-31 hold K 16-31, for row M = lane%16.
    for (int r = 0; r < 16; ++r) {
        const float* zrow = z + (size_t)(m0 + r) * CODE_DIM;
        #pragma unroll
        for (int h = 0; h < 2; ++h) {
            int kb = h * 128 + lane * 4;       // 4 consecutive k, same 8-block
            float4 v = *(const float4*)(zrow + kb);
            int c    = kb >> 5;
            int kk   = kb & 31;
            int sub  = kk >> 3;
            int Ld   = r + ((sub >> 1) << 4);
            int elem = (kk & 7) + ((sub & 1) << 3);
            int off  = c * 512 + Ld * 16 + elem;
            float xs[4] = {v.x, v.y, v.z, v.w};
            v4us hv, lv;
            #pragma unroll
            for (int j = 0; j < 4; ++j) {
                unsigned short hi, lo;
                bf16_split(xs[j], hi, lo);
                hv[j] = hi; lv[j] = lo;
            }
            *(v4us*)(aw + off)        = hv;    // ds_store_b64
            *(v4us*)(aw + 4096 + off) = lv;
        }
    }
    // same-wave DS ops are in-order; compiler inserts s_wait_dscnt for the reads.

    float bestD[8];
    int   bestI[8];
    #pragma unroll
    for (int r = 0; r < 8; ++r) { bestD[r] = 3.4e38f; bestI[r] = 0; }

    const int colLane = lane & 15;
    const int halfSel = lane >> 4;

    for (int n = 0; n < 64; ++n) {
        const int col = n * 16 + colLane;      // this lane's codebook column
        float en = enorm[col];
        const unsigned short* bhp = ebh + (size_t)col * CODE_DIM + halfSel * 16;
        const unsigned short* blp = ebl + (size_t)col * CODE_DIM + halfSel * 16;

        v8f acc;
        #pragma unroll
        for (int i = 0; i < 8; ++i) acc[i] = 0.f;

        #pragma unroll
        for (int c = 0; c < 8; ++c) {
            v16bf ah = __builtin_bit_cast(v16bf, *(const v16us*)(aw + c * 512 + lane * 16));
            v16bf al = __builtin_bit_cast(v16bf, *(const v16us*)(aw + 4096 + c * 512 + lane * 16));
            v16bf bh = __builtin_bit_cast(v16bf, *(const v16us*)(bhp + c * 32));
            v16bf bl = __builtin_bit_cast(v16bf, *(const v16us*)(blp + c * 32));
            // error-compensated bf16 product: hi*hi + hi*lo + lo*hi (f32 accumulate)
            acc = __builtin_amdgcn_wmma_f32_16x16x32_bf16(false, ah, false, bh, (short)0, acc, false, false);
            acc = __builtin_amdgcn_wmma_f32_16x16x32_bf16(false, ah, false, bl, (short)0, acc, false, false);
            acc = __builtin_amdgcn_wmma_f32_16x16x32_bf16(false, al, false, bh, (short)0, acc, false, false);
        }
        // dist = ||e||^2 - 2 z.e  (||z||^2 is row-constant, irrelevant for argmin)
        #pragma unroll
        for (int r = 0; r < 8; ++r) {
            float d = en - 2.0f * acc[r];
            if (d < bestD[r]) { bestD[r] = d; bestI[r] = col; }  // strict < keeps lowest col
        }
    }

    // ---- cross-lane argmin over 16 columns (per half-wave; row M = r + 8*halfSel) ----
    #pragma unroll
    for (int r = 0; r < 8; ++r) {
        float d = bestD[r];
        int   i = bestI[r];
        #pragma unroll
        for (int m = 1; m <= 8; m <<= 1) {
            float od = __shfl_xor(d, m, 32);
            int   oi = __shfl_xor(i, m, 32);
            if (od < d || (od == d && oi < i)) { d = od; i = oi; }
        }
        if (colLane == 0) {
            int row = m0 + r + (halfSel << 3);
            idx_out[row]   = i;
            idx_out_f[row] = (float)i;
        }
    }
}

// ---------------- kernel 4: gather quantized + scatter dw/counts + loss ----------------
__global__ void gather_scatter_kernel(const float* __restrict__ z,
                                      const float* __restrict__ embed,
                                      const int*   __restrict__ idx,
                                      float* __restrict__ qout,
                                      float* __restrict__ dw,
                                      float* __restrict__ counts,
                                      float* __restrict__ lossAcc) {
    const int m = blockIdx.x;
    const int d = threadIdx.x;
    const int k = idx[m];
    float zv = z[(size_t)m * CODE_DIM + d];
    float q  = embed[(size_t)k * CODE_DIM + d];
    qout[(size_t)m * CODE_DIM + d] = q;               // quantized_st == quantized numerically
    atomicAdd(&dw[(size_t)k * CODE_DIM + d], zv);     // segment_sum of z rows
    float diff = zv - q;
    float s = diff * diff;
    for (int msk = 16; msk; msk >>= 1) s += __shfl_xor(s, msk, 32);
    __shared__ float parts[8];
    if ((threadIdx.x & 31) == 0) parts[threadIdx.x >> 5] = s;
    __syncthreads();
    if (threadIdx.x == 0) {
        float t = 0.f;
        #pragma unroll
        for (int i = 0; i < 8; ++i) t += parts[i];
        atomicAdd(lossAcc, t);
        atomicAdd(&counts[k], 1.0f);
    }
}

// ---------------- kernel 5: EMA cluster sizes + n + loss ----------------
__global__ void finalize_cs_kernel(const float* __restrict__ ema_cs,
                                   const float* __restrict__ counts,
                                   const float* __restrict__ lossSum,
                                   float* __restrict__ out_loss,
                                   float* __restrict__ out_cs,
                                   float* __restrict__ nSum) {
    const int t = threadIdx.x;               // 1024 threads
    float ncs = ema_cs[t] * DECAY + ONE_M_D * counts[t];
    out_cs[t] = ncs;
    float s = ncs;
    for (int m = 16; m; m >>= 1) s += __shfl_xor(s, m, 32);
    __shared__ float parts[32];
    if ((t & 31) == 0) parts[t >> 5] = s;
    __syncthreads();
    if (t < 32) {
        float v = parts[t];
        for (int m = 16; m; m >>= 1) v += __shfl_xor(v, m, 32);
        if (t == 0) {
            *nSum = v;
            *out_loss = COMMIT * (*lossSum) / ((float)N_TOKENS * (float)CODE_DIM);
        }
    }
}

// ---------------- kernel 6: new_ema_w + new_embed ----------------
__global__ void finalize_embed_kernel(const float* __restrict__ ema_w,
                                      const float* __restrict__ dw,
                                      const float* __restrict__ out_cs,
                                      const float* __restrict__ nSum,
                                      float* __restrict__ out_new_embed,
                                      float* __restrict__ out_new_ema_w) {
    const int k = blockIdx.x, d = threadIdx.x;
    size_t off = (size_t)k * CODE_DIM + d;
    float nw = ema_w[off] * DECAY + ONE_M_D * dw[off];
    out_new_ema_w[off] = nw;
    float n   = *nSum;
    float ncs = out_cs[k];
    float cs  = (ncs + EPSV) / (n + (float)K_CODES * EPSV) * n;
    out_new_embed[off] = nw / cs;
}

// ---------------- launcher ----------------
extern "C" void kernel_launch(void* const* d_in, const int* in_sizes, int n_in,
                              void* d_out, int out_size, void* d_ws, size_t ws_size,
                              hipStream_t stream) {
    const float* z       = (const float*)d_in[0];   // [N, D]
    const float* embed   = (const float*)d_in[1];   // [K, D]
    const float* ema_cs  = (const float*)d_in[2];   // [K]
    const float* ema_w   = (const float*)d_in[3];   // [K, D]

    char* ws = (char*)d_ws;
    float*          dw      = (float*)(ws + WS_DW);
    float*          counts  = (float*)(ws + WS_COUNTS);
    float*          lossSum = (float*)(ws + WS_LOSS);
    float*          nSum    = (float*)(ws + WS_NSUM);
    int*            idx     = (int*)(ws + WS_IDX);
    unsigned short* ebh     = (unsigned short*)(ws + WS_EBH);
    unsigned short* ebl     = (unsigned short*)(ws + WS_EBL);
    float*          enorm   = (float*)(ws + WS_ENORM);

    // d_out tuple layout: loss[1], quantized_st[N*D], indices[N],
    //                     new_embed[K*D], new_cluster_size[K], new_ema_w[K*D]
    float* out      = (float*)d_out;
    float* o_loss   = out;
    float* o_q      = out + 1;
    float* o_idx    = o_q + (size_t)N_TOKENS * CODE_DIM;
    float* o_emb    = o_idx + N_TOKENS;
    float* o_cs     = o_emb + (size_t)K_CODES * CODE_DIM;
    float* o_emaw   = o_cs + K_CODES;

    // 1) zero dw/counts/scalars (determinism across replays)
    zero_ws_kernel<<<(ZERO_FLOATS + 255) / 256, 256, 0, stream>>>((float*)ws, ZERO_FLOATS);
    // 2) codebook bf16 hi/lo split + norms
    prep_embed_kernel<<<K_CODES, CODE_DIM, 0, stream>>>(embed, ebh, ebl, enorm);
    // 3) WMMA distances + argmin (8192 token tiles, 4 waves/block)
    vq_argmin_kernel<<<N_TOKENS / 16 / 4, 128, 0, stream>>>(z, ebh, ebl, enorm, idx, o_idx);
    // 4) gather + scatter + loss
    gather_scatter_kernel<<<N_TOKENS, CODE_DIM, 0, stream>>>(z, embed, idx, o_q, dw, counts, lossSum);
    // 5) EMA cluster sizes, n, loss
    finalize_cs_kernel<<<1, K_CODES, 0, stream>>>(ema_cs, counts, lossSum, o_loss, o_cs, nSum);
    // 6) new_ema_w / new_embed
    finalize_embed_kernel<<<K_CODES, CODE_DIM, 0, stream>>>(ema_w, dw, o_cs, nSum, o_emb, o_emaw);
}
